// RayTracing_73392401154618
// MI455X (gfx1250) — matrix operand order repaired and checked by reading, not compiled, not run on hardware
//
#include <hip/hip_runtime.h>
#include <hip/hip_bf16.h>
#include <math.h>

// ---------------------------------------------------------------------------
// CDNA5 (gfx1250) IDR-style sphere-tracing ray sampler.
//
//  * Layer-1 latent fold: c1 = b1 + id@W1[3:35] + exp@W1[35:67]  (prologue)
//  * 64x64 hidden layer via V_WMMA_F32_16X16X32_F16, 32 points per wave
//  * W2 pre-swizzled into B-fragment layout, scaled by ln2, staged in LDS
//  * All softplus in log2 domain: u = log2(1+exp2(y)), exact for x < 88.7
//    (raw v_exp_f32/v_log_f32, guard-free: log arg >= 1, no denorm path)
//  * Layer-3: register-accumulated partial dot per C tile, then a single
//    LDS transpose round-trip (padded rows) to redistribute per-point sums
//  * All reference masking via selects -> EXEC stays full at WMMA sites
// ---------------------------------------------------------------------------

typedef __attribute__((ext_vector_type(16))) _Float16 v16h;
typedef __attribute__((ext_vector_type(8)))  float    v8f;

#define TH_SDF  5e-5f
#define NSTEPS  100
#define BLOCK   128
#define LOG2E   1.44269504088896f
#define LN2     0.69314718055995f
#define PROW    20                      // padded row stride (words)

// softplus/ln2 with log2-domain argument y = x*log2e.
// Exact for y < 128 (x < 88.7); the reference's log1p(exp(x)) overflows
// at the same point in f32. Argument of log2 is >= 1: no denorm fixups.
__device__ __forceinline__ float sp_u(float y) {
  float e = __builtin_amdgcn_exp2f(y);
  return __builtin_amdgcn_logf(1.0f + e);
}

__device__ __forceinline__ float fast_tanh(float x) {
  x = fminf(fmaxf(x, -15.0f), 15.0f);
  float e = __builtin_amdgcn_exp2f(x * (2.0f * LOG2E));
  return (e - 1.0f) * __builtin_amdgcn_rcpf(e + 1.0f);
}

// Batched SDF: evaluates sdf at (px,py,pz) for all 32 lanes of the wave.
__device__ __forceinline__ float sdf_eval(
    float px, float py, float pz,
    const v16h*  s_w2f,                // [256] LDS: B fragments of ln2*W2
    const float4* __restrict__ s_l1,   // [64]: log2e*(w1x, w1y, w1z, c1)
    const float2* __restrict__ s_bw,   // [64]: (log2e*b2, w3)
    float* s_part,                     // [32*PROW] per-wave transpose pad
    float b3)
{
  const int lane = threadIdx.x & 31;
  const int g    = lane >> 4;          // half-wave group
  const int col  = lane & 15;

  #pragma unroll
  for (int rt = 0; rt < 2; ++rt) {
    // Broadcast coordinates of the 16 points in this row tile.
    int src = col + 16 * rt;
    float qx = __shfl(px, src, 32);
    float qy = __shfl(py, src, 32);
    float qz = __shfl(pz, src, 32);

    // Layer 1 (3x64 + folded bias) straight into the A-fragment layout
    // of V_WMMA_F32_16X16X32_F16 (ISA 7.12.2, 16-bit A 16x32).
    v16h a0, a1;
    #pragma unroll
    for (int e = 0; e < 16; ++e) {
      int kk = (e < 8) ? (8 * g + e) : (16 + 8 * g + (e - 8));
      float4 w0 = s_l1[kk];
      float4 w1 = s_l1[32 + kk];
      float u0 = sp_u(fmaf(qx, w0.x, fmaf(qy, w0.y, fmaf(qz, w0.z, w0.w))));
      float u1 = sp_u(fmaf(qx, w1.x, fmaf(qy, w1.y, fmaf(qz, w1.z, w1.w))));
      a0[e] = (_Float16)u0;
      a1[e] = (_Float16)u1;
    }

    // Layer 2 (64x64) via WMMA (W2 pre-scaled by ln2 compensates the
    // log2-domain A fragment); layer-3 partials accumulate in registers.
    float part[8];
    #pragma unroll
    for (int r = 0; r < 8; ++r) part[r] = 0.0f;

    #pragma unroll
    for (int nt = 0; nt < 4; ++nt) {
      float2 bw = s_bw[16 * nt + col];  // (log2e*b2[n], w3[n]), n=16*nt+col
      v16h b0 = s_w2f[nt * 32 + lane];
      v16h b1 = s_w2f[(4 + nt) * 32 + lane];
      v8f c = {};                        // SRC2 = inline 0
      c = __builtin_amdgcn_wmma_f32_16x16x32_f16(false, a0, false, b0,
                                                 (short)0, c, false, false);
      c = __builtin_amdgcn_wmma_f32_16x16x32_f16(false, a1, false, b1,
                                                 (short)0, c, false, false);
      #pragma unroll
      for (int r = 0; r < 8; ++r) {
        float u2 = sp_u(fmaf(c[r], LOG2E, bw.x));
        part[r] = fmaf(u2, bw.y, part[r]);
      }
    }

    // One store pass into the transpose pad (rows p and p+8 in disjoint
    // bank sets thanks to the 20-word row stride).
    #pragma unroll
    for (int r = 0; r < 8; ++r)
      s_part[(16 * rt + 8 * g + r) * PROW + col] = part[r];
  }
  asm volatile("s_wait_dscnt 0" ::: "memory");   // intra-wave LDS ordering
  // Each lane sums its own point's 16 column-partials (transpose read).
  const float4* row4 = (const float4*)(s_part + lane * PROW);
  float4 x0 = row4[0], x1 = row4[1], x2 = row4[2], x3 = row4[3];
  float sum = (((x0.x + x0.y) + (x0.z + x0.w)) + ((x1.x + x1.y) + (x1.z + x1.w)))
            + (((x2.x + x2.y) + (x2.z + x2.w)) + ((x3.x + x3.y) + (x3.z + x3.w)));
  asm volatile("" ::: "memory");                 // keep next eval's stores after
  float delta = sum * LN2;                       // ln2 factored out of the dot

  float r2 = fmaf(px, px, fmaf(py, py, fmaf(pz, pz, 1e-12f)));
  return sqrtf(r2) - 0.5f + 0.05f * fast_tanh(delta + b3);
}

// ---------------------------------------------------------------------------
// Prologue: fold latents into c1 and pre-swizzle ln2*W2 into the B-fragment
// layout. B fragment (32x16 f16): element e of lane half h holds K=8h+e (e<8),
// K=16+8h+(e-8) (e>=8); N = lane&15 (+ tile offsets).
// ---------------------------------------------------------------------------
__global__ void rt_prologue(
    const float* __restrict__ idl, const float* __restrict__ expl,
    const float* __restrict__ W1,  const float* __restrict__ b1,
    const float* __restrict__ W2,
    float* __restrict__ ws_c1, _Float16* __restrict__ ws_w2h)
{
  int tid = threadIdx.x;
  if (tid < 64) {
    float acc = b1[tid];
    for (int j = 0; j < 32; ++j) acc = fmaf(idl[j],  W1[(3  + j) * 64 + tid], acc);
    for (int j = 0; j < 32; ++j) acc = fmaf(expl[j], W1[(35 + j) * 64 + tid], acc);
    ws_c1[tid] = acc;
  }
  int f    = tid >> 5;               // fragment id 0..7 (kt = f>>2, nt = f&3)
  int lane = tid & 31;
  int kt = f >> 2, nt = f & 3;
  int g = lane >> 4, col = lane & 15;
  _Float16* dst = ws_w2h + (f * 32 + lane) * 16;
  #pragma unroll
  for (int e = 0; e < 16; ++e) {
    int row = 32 * kt + ((e < 8) ? (8 * g + e) : (16 + 8 * g + (e - 8)));  // K
    dst[e] = (_Float16)(LN2 * W2[row * 64 + 16 * nt + col]);
  }
}

// ---------------------------------------------------------------------------
// Main kernel: one ray per lane, full reference algorithm, batched SDF.
// ---------------------------------------------------------------------------
__global__ __launch_bounds__(BLOCK) void rt_main(
    const float* __restrict__ cam,
    const int*   __restrict__ objmask,
    const float* __restrict__ dirs,
    const float* __restrict__ W1,
    const float* __restrict__ b2g,
    const float* __restrict__ W3g,
    const float* __restrict__ b3g,
    const float* __restrict__ ws_c1,
    const v16h*  __restrict__ ws_w2,
    float* __restrict__ out,
    int n)
{
  __shared__ v16h   s_w2f[256];                        // 8 KB: W2 B-fragments
  __shared__ float4 s_l1[64];
  __shared__ float2 s_bw[64];
  __shared__ float4 s_part_all[(BLOCK / 32) * 8 * PROW]; // 32*PROW floats/wave

  int tid = threadIdx.x;
  for (int i = tid; i < 256; i += BLOCK) s_w2f[i] = ws_w2[i];
  if (tid < 64) {
    s_l1[tid] = make_float4(LOG2E * W1[tid], LOG2E * W1[64 + tid],
                            LOG2E * W1[128 + tid], LOG2E * ws_c1[tid]);
    s_bw[tid] = make_float2(LOG2E * b2g[tid], W3g[tid]);
  }
  __syncthreads();

  int lane = tid & 31;
  float* s_part = (float*)(s_part_all + (tid >> 5) * 8 * PROW);

  float b3 = b3g[0];
  int ray = blockIdx.x * BLOCK + tid;
  float cx = cam[0], cy = cam[1], cz = cam[2];
  float dx = dirs[ray * 3], dy = dirs[ray * 3 + 1], dz = dirs[ray * 3 + 2];
  bool omask = objmask[ray] != 0;

  // ---- bounding-sphere intersection (R = 1) ----
  float rcd   = dx * cx + dy * cy + dz * cz;
  float c2    = cx * cx + cy * cy + cz * cz;
  float under = rcd * rcd - (c2 - 1.0f);
  bool  mi    = under > 0.0f;
  float sq    = sqrtf(mi ? under : 1.0f);
  float i0    = mi ? fmaxf(-sq - rcd, 0.0f) : 0.0f;
  float i1    = mi ? fmaxf( sq - rcd, 0.0f) : 0.0f;

  // ---- sphere tracing ----
  bool unfin_s = mi, unfin_e = mi;
  float acc_s = i0, acc_e = i1;
  float psx = mi ? fmaf(acc_s, dx, cx) : 0.0f;
  float psy = mi ? fmaf(acc_s, dy, cy) : 0.0f;
  float psz = mi ? fmaf(acc_s, dz, cz) : 0.0f;
  float pex = mi ? fmaf(acc_e, dx, cx) : 0.0f;
  float pey = mi ? fmaf(acc_e, dy, cy) : 0.0f;
  float pez = mi ? fmaf(acc_e, dz, cz) : 0.0f;
  float v;
  v = sdf_eval(psx, psy, psz, s_w2f, s_l1, s_bw, s_part, b3);
  float next_s = unfin_s ? v : 0.0f;
  v = sdf_eval(pex, pey, pez, s_w2f, s_l1, s_bw, s_part, b3);
  float next_e = unfin_e ? v : 0.0f;
  bool no_inter = (next_e < 0.0f) && (next_s < 0.0f);

  #pragma unroll 1
  for (int it = 0; it <= 10; ++it) {
    float cs = unfin_s ? next_s : 0.0f; cs = (cs <= TH_SDF) ? 0.0f : cs;
    float ce = unfin_e ? next_e : 0.0f; ce = (ce <= TH_SDF) ? 0.0f : ce;
    unfin_s = unfin_s && (cs > TH_SDF);
    unfin_e = unfin_e && (ce > TH_SDF);
    if (it == 10) break;
    cs = fminf(fmaxf(cs, -0.5f), 0.5f);
    ce = fminf(fmaxf(ce, -0.5f), 0.5f);
    acc_s = fmaf( 1.2f, cs, acc_s);
    acc_e = fmaf(-1.2f, ce, acc_e);
    psx = fmaf(acc_s, dx, cx); psy = fmaf(acc_s, dy, cy); psz = fmaf(acc_s, dz, cz);
    pex = fmaf(acc_e, dx, cx); pey = fmaf(acc_e, dy, cy); pez = fmaf(acc_e, dz, cz);
    v = sdf_eval(psx, psy, psz, s_w2f, s_l1, s_bw, s_part, b3);
    next_s = unfin_s ? v : 0.0f;
    v = sdf_eval(pex, pey, pez, s_w2f, s_l1, s_bw, s_part, b3);
    next_e = unfin_e ? v : 0.0f;
    bool np_s = next_s < 0.0f, np_e = next_e < 0.0f;
    // line search (1 iter, step 0.5)
    acc_s = np_s ? (acc_s - 0.5f * cs) : acc_s;
    acc_e = np_e ? (acc_e + 0.5f * ce) : acc_e;
    psx = fmaf(acc_s, dx, cx); psy = fmaf(acc_s, dy, cy); psz = fmaf(acc_s, dz, cz);
    pex = fmaf(acc_e, dx, cx); pey = fmaf(acc_e, dy, cy); pez = fmaf(acc_e, dz, cz);
    v = sdf_eval(psx, psy, psz, s_w2f, s_l1, s_bw, s_part, b3);
    next_s = np_s ? v : next_s;
    v = sdf_eval(pex, pey, pez, s_w2f, s_l1, s_bw, s_part, b3);
    next_e = np_e ? v : next_e;
    bool inside = acc_s < acc_e;
    unfin_s = unfin_s && inside;
    unfin_e = unfin_e && inside;
  }

  bool net_obj_mask = (acc_s < acc_e) && (!no_inter);
  bool smask = unfin_s;

  // ---- ray sampler: 100 uniform samples, rolling argmins ----
  float mn = smask ? acc_s : 0.0f;
  float mx = smask ? acc_e : 0.0f;
  float span = mx - mn;

  float bestKey = 3.4e38f, bestSdf = 3.4e38f;
  int   ind = 0;
  float d_ind = 0.f, sdf_ind = 0.f, lo_d = 0.f, lo_sdf = 0.f, d_omin = 0.f;
  float prev_d = 0.f, prev_sdf = 0.f;

  #pragma unroll 1
  for (int s = 0; s < NSTEPS; ++s) {
    float t    = (float)s * (1.0f / 99.0f);
    float dcur = fmaf(t, span, mn);
    float qx = fmaf(dcur, dx, cx), qy = fmaf(dcur, dy, cy), qz = fmaf(dcur, dz, cz);
    v = sdf_eval(qx, qy, qz, s_w2f, s_l1, s_bw, s_part, b3);
    float sgn = (v > 0.f) ? 1.f : ((v < 0.f) ? -1.f : 0.f);
    float key = sgn * (float)(NSTEPS - s);
    bool upd = key < bestKey;           // first-occurrence argmin
    bestKey = upd ? key  : bestKey;
    ind     = upd ? s    : ind;
    d_ind   = upd ? dcur : d_ind;
    sdf_ind = upd ? v    : sdf_ind;
    lo_d    = upd ? prev_d   : lo_d;
    lo_sdf  = upd ? prev_sdf : lo_sdf;
    bool updo = v < bestSdf;
    bestSdf = updo ? v    : bestSdf;
    d_omin  = updo ? dcur : d_omin;
    prev_d = dcur; prev_sdf = v;
  }
  if (ind == 0) { lo_d = prev_d; lo_sdf = prev_sdf; }   // (ind-1) % 100 == 99

  bool net_surf = sdf_ind < 0.0f;
  float samp_d  = d_ind;
  bool p_out    = !(omask && net_surf);
  samp_d = p_out ? d_omin : samp_d;
  bool samp_net_obj = smask && net_surf;
  bool secant = smask && net_surf && omask;

  // ---- secant refinement (8 iters) ----
  float z_high   = secant ? d_ind   :  1.0f;
  float sdf_high = secant ? sdf_ind : -1.0f;
  float z_low    = secant ? lo_d    :  0.0f;
  float sdf_low  = secant ? lo_sdf  :  1.0f;
  float denom = sdf_high - sdf_low;
  denom = (denom == 0.0f) ? 1.0f : denom;
  float z_pred = fmaf(-sdf_low, (z_high - z_low) / denom, z_low);

  #pragma unroll 1
  for (int si = 0; si < 8; ++si) {
    float qx = fmaf(z_pred, dx, cx), qy = fmaf(z_pred, dy, cy), qz = fmaf(z_pred, dz, cz);
    v = sdf_eval(qx, qy, qz, s_w2f, s_l1, s_bw, s_part, b3);
    bool glo = v > 0.0f, ghi = v < 0.0f;
    z_low  = glo ? z_pred : z_low;   sdf_low  = glo ? v : sdf_low;
    z_high = ghi ? z_pred : z_high;  sdf_high = ghi ? v : sdf_high;
    denom = sdf_high - sdf_low;
    denom = (denom == 0.0f) ? 1.0f : denom;
    z_pred = fmaf(-sdf_low, (z_high - z_low) / denom, z_low);
  }
  samp_d = secant ? z_pred : samp_d;
  float spx = fmaf(samp_d, dx, cx), spy = fmaf(samp_d, dy, cy), spz = fmaf(samp_d, dz, cz);

  // ---- final outputs ----
  float ox = smask ? spx : psx;
  float oy = smask ? spy : psy;
  float oz = smask ? spz : psz;
  float od = smask ? samp_d : acc_s;
  bool  om = smask ? samp_net_obj : net_obj_mask;

  out[ray * 3 + 0] = ox;
  out[ray * 3 + 1] = oy;
  out[ray * 3 + 2] = oz;
  out[3 * n + ray] = om ? 1.0f : 0.0f;
  out[4 * n + ray] = od;
}

extern "C" void kernel_launch(void* const* d_in, const int* in_sizes, int n_in,
                              void* d_out, int out_size, void* d_ws, size_t ws_size,
                              hipStream_t stream) {
  const float* cam  = (const float*)d_in[0];
  const int*   om   = (const int*)  d_in[1];
  const float* dirs = (const float*)d_in[2];
  const float* idl  = (const float*)d_in[3];
  const float* expl = (const float*)d_in[4];
  const float* W1   = (const float*)d_in[5];
  const float* b1   = (const float*)d_in[6];
  const float* W2   = (const float*)d_in[7];
  const float* b2   = (const float*)d_in[8];
  const float* W3   = (const float*)d_in[9];
  const float* b3   = (const float*)d_in[10];
  float* out = (float*)d_out;
  int n = in_sizes[1];                 // B*P rays (16384)

  float*    ws_c1  = (float*)d_ws;
  _Float16* ws_w2h = (_Float16*)((char*)d_ws + 256);

  rt_prologue<<<1, 256, 0, stream>>>(idl, expl, W1, b1, W2, ws_c1, ws_w2h);
  int grid = (n + BLOCK - 1) / BLOCK;
  rt_main<<<grid, BLOCK, 0, stream>>>(cam, om, dirs, W1, b2, W3, b3,
                                      ws_c1, (const v16h*)ws_w2h, out, n);
}
